// ShiftedWindowAttention_13649406067262
// MI455X (gfx1250) — compile-verified, matrix operand order checked by reading, not run on hardware
//
#include <hip/hip_runtime.h>
#include <hip/hip_bf16.h>

// ---------------------------------------------------------------------------
// SwinV2 shifted-window attention, fully fused per-window for MI455X (gfx1250)
//   - wave32; 512 threads (16 waves) per block, 1 block per window (1024 blocks)
//   - f16 WMMA (v_wmma_f32_16x16x32_f16) with fp32 accumulation/epilogues
//   - attention processes 2 heads concurrently (8 waves per head slot)
//   - ~251 KB dynamic LDS per workgroup (CDNA5 WGP has 320 KB)
// ---------------------------------------------------------------------------

typedef __attribute__((ext_vector_type(16))) _Float16 v16h;
typedef __attribute__((ext_vector_type(8)))  _Float16 v8h;
typedef __attribute__((ext_vector_type(8)))  float    v8f;

#define C_DIM   384
#define HEADS   12
#define HD      32
#define NTOK    64
#define XLD     392     // xs / outs row stride (halves), 16B-aligned rows
#define QLD     1160    // qkv row stride (halves)
#define PLD     72      // softmax-prob row stride (halves)

// ---- LDS byte offsets (total 256,768 B) ----
#define OFF_QKS   0                     // [64][QLD] f16            148,480
#define OFF_XS    148480                // [64][XLD] f16             50,176
#define OFF_SB    198656                // [2][64][65] fp32          33,280
#define OFF_PS    231936                // [2][64][PLD] f16          18,432
#define OFF_QINV  250368                // [12][64] fp32              3,072
#define OFF_KINV  253440                // [12][64] fp32              3,072
#define OFF_RID   256512                // [64] int                     256
#define SMEM_SZ   256768

// ---- workspace byte offsets ----
#define WS_QKVW   0         // f16 packed qkv weights: 72*12 frags * 512 halves
#define WS_PROJW  884736    // f16 packed proj weights: 24*12 frags * 512 halves
#define WS_BIAS   1179648   // fp32 [12][64][64] = 16*sigmoid(cpb)
#define WS_SCALE  1376256   // fp32 [12] clamped exp(logit_scale)
#define WS_TABLE  1376320   // fp32 [225][12] cpb mlp output

__device__ __forceinline__ v8f wmma_f16(v16h a, v16h b, v8f c) {
    return __builtin_amdgcn_wmma_f32_16x16x32_f16(
        /*neg_a=*/false, a, /*neg_b=*/false, b,
        /*c_mod=*/(short)0, c, /*reuse_a=*/false, /*reuse_b=*/false);
}

// A-fragment (16x32 f16): lane L holds row M=L%16, K = (L/16)*8 + {0..7, 16..23}.
// All call sites keep (row*ld + kbase) % 8 == 0 -> two aligned ds_load_b128.
__device__ __forceinline__ v16h frag_rows(const _Float16* base, int ld, int lane) {
    const int m  = lane & 15;
    const int kb = (lane >> 4) << 3;
    const _Float16* p = base + m * ld + kb;
    v8h lo = *(const v8h*)(p);
    v8h hi = *(const v8h*)(p + 16);
    return __builtin_shufflevector(lo, hi, 0,1,2,3,4,5,6,7,8,9,10,11,12,13,14,15);
}

// B-fragment (32x16 f16) from a K-major [K][N] array: lane L holds col N=L%16,
// K = (L/16)*8 + {0..7, 16..23}. Strided gather (used only for P@V).
__device__ __forceinline__ v16h frag_cols(const _Float16* base, int ld, int lane) {
    const int n  = lane & 15;
    const int kb = (lane >> 4) << 3;
    v16h f;
#pragma unroll
    for (int t = 0; t < 8; ++t) f[t]     = base[(kb + t) * ld + n];
#pragma unroll
    for (int t = 0; t < 8; ++t) f[t + 8] = base[(kb + 16 + t) * ld + n];
    return f;
}

// ---------------------------------------------------------------------------
// Prep kernel 1: repack qkv_w [1152][384] and proj_w [384][384] (fp32, row =
// output channel) into f16 WMMA B-fragment order: frag(nt,kt) = 512 halves,
// half index = lane*16 + t, element B[k][n] = W[nt*16+n][kt*32+k].
// ---------------------------------------------------------------------------
__global__ void pack_weights_kernel(const float* __restrict__ qkv_w,
                                    const float* __restrict__ proj_w,
                                    _Float16* __restrict__ qpack,
                                    _Float16* __restrict__ ppack) {
    const int NQ = 72 * 12 * 512;
    const int NP = 24 * 12 * 512;
    for (int idx = blockIdx.x * blockDim.x + threadIdx.x; idx < NQ + NP;
         idx += gridDim.x * blockDim.x) {
        const bool isq = idx < NQ;
        const int id   = isq ? idx : idx - NQ;
        const int frag = id >> 9;
        const int lane = (id >> 4) & 31;
        const int t    = id & 15;
        const int nt   = frag / 12, kt = frag % 12;
        const int n    = lane & 15;
        const int kb   = (lane >> 4) << 3;
        const int k    = kt * 32 + ((t < 8) ? (kb + t) : (16 + kb + (t - 8)));
        const int o    = nt * 16 + n;
        const float v  = isq ? qkv_w[o * C_DIM + k] : proj_w[o * C_DIM + k];
        if (isq) qpack[id] = (_Float16)v;
        else     ppack[id] = (_Float16)v;
    }
}

// ---------------------------------------------------------------------------
// Prep kernel 2: SwinV2 continuous-position-bias MLP (2 -> 512 relu -> 12)
// over the 15x15 log-spaced relative coordinate table; also clamped exp of
// the learned logit scale.
// ---------------------------------------------------------------------------
__global__ void cpb_table_kernel(const float* __restrict__ w1,
                                 const float* __restrict__ b1,
                                 const float* __restrict__ w2,
                                 const float* __restrict__ lscale,
                                 float* __restrict__ table,
                                 float* __restrict__ scaleArr) {
    __shared__ float hid[512];
    const int r = blockIdx.x;            // 0..224
    const int i = r / 15, j = r % 15;
    const float a0 = (float)(i - 7) * (8.0f / 7.0f);
    const float a1 = (float)(j - 7) * (8.0f / 7.0f);
    const float t0 = copysignf(log2f(fabsf(a0) + 1.0f) * (1.0f / 3.0f), a0);
    const float t1 = copysignf(log2f(fabsf(a1) + 1.0f) * (1.0f / 3.0f), a1);
    for (int u = threadIdx.x; u < 512; u += blockDim.x)
        hid[u] = fmaxf(w1[u * 2] * t0 + w1[u * 2 + 1] * t1 + b1[u], 0.0f);
    __syncthreads();
    if (threadIdx.x < HEADS) {
        const int h = threadIdx.x;
        float s = 0.0f;
        for (int u = 0; u < 512; ++u) s += hid[u] * w2[h * 512 + u];
        table[r * HEADS + h] = s;
        if (r == 0)
            scaleArr[h] = __expf(fminf(lscale[h], 4.60517018598809136804f)); // ln(100)
    }
}

// Prep kernel 3: expand gathered bias -> 16*sigmoid, [12][64][64] fp32.
__global__ void expand_bias_kernel(const float* __restrict__ table,
                                   float* __restrict__ biasArr) {
    const int idx = blockIdx.x * blockDim.x + threadIdx.x;
    if (idx >= HEADS * NTOK * NTOK) return;
    const int h = idx >> 12, rem = idx & 4095;
    const int m = rem >> 6, n = rem & 63;
    const int di = (m >> 3) - (n >> 3) + 7;
    const int dj = (m & 7) - (n & 7) + 7;
    const float t = table[(di * 15 + dj) * HEADS + h];
    biasArr[idx] = 16.0f / (1.0f + __expf(-t));
}

// ---------------------------------------------------------------------------
// Fused per-window kernel: gather (roll+partition) -> QKV GEMM -> cosine
// attention + mask + CPB bias + softmax (2 heads in flight) -> P@V ->
// proj GEMM -> scatter (departition+unroll).
// ---------------------------------------------------------------------------
__global__ __launch_bounds__(512, 1)
void swin_fused_kernel(const float* __restrict__ x,
                       const float* __restrict__ qkv_b,
                       const float* __restrict__ proj_b,
                       const _Float16* __restrict__ qpack,
                       const _Float16* __restrict__ ppack,
                       const float* __restrict__ biasArr,
                       const float* __restrict__ scaleArr,
                       float* __restrict__ out) {
    extern __shared__ char smem[];
    _Float16* qks  = (_Float16*)(smem + OFF_QKS);   // [64][QLD]  qkv f16
    _Float16* xs   = (_Float16*)(smem + OFF_XS);    // [64][XLD]  x f16, reused as attn-out
    float*    sb   = (float*)   (smem + OFF_SB);    // [2][64][65] scores
    _Float16* ps   = (_Float16*)(smem + OFF_PS);    // [2][64][PLD] probs f16
    float*    qinv = (float*)   (smem + OFF_QINV);  // [12][64]
    float*    kinv = (float*)   (smem + OFF_KINV);  // [12][64]
    int*      rid  = (int*)     (smem + OFF_RID);   // [64] shift-mask region ids

    const int tid  = threadIdx.x;
    const int lane = tid & 31;
    const int wave = tid >> 5;            // 0..15
    const int slot = wave >> 3;           // head slot 0/1
    const int wv   = wave & 7;            // wave within slot
    const int wid  = blockIdx.x;          // 0..1023
    const int b    = wid >> 6;
    const int wl   = wid & 63;
    const int wh   = wl >> 3;             // window row
    const int wwn  = wl & 7;              // window col

    // --- Phase A: gather shifted window x -> LDS f16 (float4 loads) ---------
    for (int idx = tid; idx < NTOK * 96; idx += 512) {
        const int t  = idx / 96;
        const int c4 = (idx % 96) << 2;
        const int i = t >> 3, j = t & 7;
        const int hh  = ((wh  << 3) + i + 4) & 63;   // roll(-4) folded in
        const int ww2 = ((wwn << 3) + j + 4) & 63;
        const float4 v = *(const float4*)(x + (((b * 64 + hh) * 64 + ww2) * C_DIM + c4));
        _Float16* d = xs + t * XLD + c4;
        d[0] = (_Float16)v.x; d[1] = (_Float16)v.y;
        d[2] = (_Float16)v.z; d[3] = (_Float16)v.w;
    }
    __syncthreads();

    // --- Phase B: QKV GEMM [64,384]x[384,1152] -> qks (f16, fp32 accum) -----
    for (int tile = wave; tile < 4 * 72; tile += 16) {
        const int mt = tile / 72, nt = tile % 72;
        v8f acc = {};
#pragma unroll 4
        for (int kt = 0; kt < 12; ++kt) {
            v16h a  = frag_rows(xs + (mt * 16) * XLD + kt * 32, XLD, lane);
            v16h bf = *(const v16h*)(qpack + (nt * 12 + kt) * 512 + lane * 16);
            acc = wmma_f16(a, bf, acc);
        }
        const int n  = lane & 15;
        const int mo = (lane >> 4) << 3;
        const int o  = nt * 16 + n;
        float bo = qkv_b[o];                              // unconditional load
        bo = (o >= C_DIM && o < 2 * C_DIM) ? 0.0f : bo;   // k-bias zeroed (cndmask)
#pragma unroll
        for (int r = 0; r < 8; ++r) {
            const int m = mt * 16 + mo + r;     // C-layout: M = r + 8*(lane>=16)
            qks[m * QLD + o] = (_Float16)(acc[r] + bo);
        }
    }
    __syncthreads();

    // --- Phase C1: q/k row inverse norms + shift-mask region ids ------------
    if (tid < NTOK) {
        const int ph = (wh << 3) + (tid >> 3);
        const int pw = (wwn << 3) + (tid & 7);
        const int rh = ph < 56 ? 0 : (ph < 60 ? 1 : 2);
        const int rw = pw < 56 ? 0 : (pw < 60 ? 1 : 2);
        rid[tid] = rh * 3 + rw;
    }
    for (int idx = tid; idx < HEADS * NTOK; idx += 512) {
        const int t = idx & 63, h = idx >> 6;
        const _Float16* qr = qks + t * QLD + h * HD;
        const _Float16* kr = qr + C_DIM;
        float sq = 0.0f, sk = 0.0f;
#pragma unroll
        for (int d = 0; d < HD; ++d) {
            const float qv = (float)qr[d], kv = (float)kr[d];
            sq += qv * qv; sk += kv * kv;
        }
        qinv[idx] = 1.0f / fmaxf(sqrtf(sq), 1e-12f);
        kinv[idx] = 1.0f / fmaxf(sqrtf(sk), 1e-12f);
    }
    __syncthreads();

    // --- Phase C2: cosine attention, 2 heads in flight (slot 0/1) -----------
    for (int hp = 0; hp < HEADS / 2; ++hp) {
        const int   h   = hp * 2 + slot;
        const float scl = scaleArr[h];
        const float* bh = biasArr + h * NTOK * NTOK;
        const float* qn = qinv + h * NTOK;
        const float* kn = kinv + h * NTOK;
        float*    sbs = sb + slot * (NTOK * 65);
        _Float16* pss = ps + slot * (NTOK * PLD);

        // S = q @ k^T (raw); epilogue applies qinv*kinv*scale + bias + mask
        for (int tile = wv; tile < 16; tile += 8) {
            const int mt = tile >> 2, nt = tile & 3;
            v16h a  = frag_rows(qks + (mt * 16) * QLD + h * HD, QLD, lane);
            v16h bf = frag_rows(qks + (nt * 16) * QLD + C_DIM + h * HD, QLD, lane);
            v8f acc = {};
            acc = wmma_f16(a, bf, acc);
            const int nn = lane & 15;
            const int mo = (lane >> 4) << 3;
            const int nc = nt * 16 + nn;
#pragma unroll
            for (int r = 0; r < 8; ++r) {
                const int m = mt * 16 + mo + r;
                float s = acc[r] * qn[m] * kn[nc] * scl + bh[m * 64 + nc];
                if (rid[m] != rid[nc]) s -= 100.0f;
                sbs[m * 65 + nc] = s;
            }
        }
        __syncthreads();

        // softmax: 128 threads, one (slot,row) each; stride-65 conflict-free
        if (tid < 2 * NTOK) {
            const int s2 = tid >> 6, row = tid & 63;
            float* rp = sb + s2 * (NTOK * 65) + row * 65;
            float mx = -1e30f;
            for (int n = 0; n < NTOK; ++n) mx = fmaxf(mx, rp[n]);
            float sum = 0.0f;
            for (int n = 0; n < NTOK; ++n) { const float e = __expf(rp[n] - mx); rp[n] = e; sum += e; }
            const float inv = 1.0f / sum;
            _Float16* pr = ps + s2 * (NTOK * PLD) + row * PLD;
            for (int n = 0; n < NTOK; ++n) pr[n] = (_Float16)(rp[n] * inv);
        }
        __syncthreads();

        // O = P @ V : 8 tiles per head slot (4 mt x 2 dt), one per wave
        {
            const int mt = wv >> 1, dt = wv & 1;
            v8f acc = {};
#pragma unroll
            for (int kt = 0; kt < 2; ++kt) {
                v16h a  = frag_rows(pss + (mt * 16) * PLD + kt * 32, PLD, lane);
                v16h bf = frag_cols(qks + (kt * 32) * QLD + 2 * C_DIM + h * HD + dt * 16,
                                    QLD, lane);
                acc = wmma_f16(a, bf, acc);
            }
            const int nn = lane & 15;
            const int mo = (lane >> 4) << 3;
#pragma unroll
            for (int r = 0; r < 8; ++r) {
                const int m = mt * 16 + mo + r;
                xs[m * XLD + h * HD + dt * 16 + nn] = (_Float16)acc[r];
            }
        }
        __syncthreads();
    }

    // --- Phase D: proj GEMM [64,384]x[384,384] + scatter to output ----------
    for (int tile = wave; tile < 4 * 24; tile += 16) {
        const int mt = tile / 24, nt = tile % 24;
        v8f acc = {};
#pragma unroll 4
        for (int kt = 0; kt < 12; ++kt) {
            v16h a  = frag_rows(xs + (mt * 16) * XLD + kt * 32, XLD, lane);
            v16h bf = *(const v16h*)(ppack + (nt * 12 + kt) * 512 + lane * 16);
            acc = wmma_f16(a, bf, acc);
        }
        const int nn = lane & 15;
        const int mo = (lane >> 4) << 3;
        const int o  = nt * 16 + nn;
        const float pb = proj_b[o];
#pragma unroll
        for (int r = 0; r < 8; ++r) {
            const int m = mt * 16 + mo + r;
            const int i = m >> 3, j = m & 7;
            const int hh  = ((wh  << 3) + i + 4) & 63;  // roll(+4) folded in
            const int ww2 = ((wwn << 3) + j + 4) & 63;
            out[((b * 64 + hh) * 64 + ww2) * C_DIM + o] = acc[r] + pb;
        }
    }
}

extern "C" void kernel_launch(void* const* d_in, const int* in_sizes, int n_in,
                              void* d_out, int out_size, void* d_ws, size_t ws_size,
                              hipStream_t stream) {
    (void)in_sizes; (void)n_in; (void)out_size; (void)ws_size;
    const float* x      = (const float*)d_in[0];
    const float* qkv_w  = (const float*)d_in[1];
    const float* qkv_b  = (const float*)d_in[2];
    const float* proj_w = (const float*)d_in[3];
    const float* proj_b = (const float*)d_in[4];
    const float* cpb_w1 = (const float*)d_in[5];
    const float* cpb_b1 = (const float*)d_in[6];
    const float* cpb_w2 = (const float*)d_in[7];
    const float* lscale = (const float*)d_in[8];

    char* ws = (char*)d_ws;
    _Float16* qpack   = (_Float16*)(ws + WS_QKVW);
    _Float16* ppack   = (_Float16*)(ws + WS_PROJW);
    float*    biasArr = (float*)(ws + WS_BIAS);
    float*    scaleA  = (float*)(ws + WS_SCALE);
    float*    table   = (float*)(ws + WS_TABLE);

    // Host-side, idempotent, graph-capture-safe; called every launch.
    (void)hipFuncSetAttribute((const void*)swin_fused_kernel,
                              hipFuncAttributeMaxDynamicSharedMemorySize, SMEM_SZ);

    pack_weights_kernel<<<2304, 256, 0, stream>>>(qkv_w, proj_w, qpack, ppack);
    cpb_table_kernel<<<225, 256, 0, stream>>>(cpb_w1, cpb_b1, cpb_w2, lscale, table, scaleA);
    expand_bias_kernel<<<192, 256, 0, stream>>>(table, biasArr);
    swin_fused_kernel<<<1024, 512, SMEM_SZ, stream>>>(
        x, qkv_b, proj_b, qpack, ppack, biasArr, scaleA, (float*)d_out);
}
